// TritonGatherConv_27711128994525
// MI455X (gfx1250) — compile-verified, matrix operand order checked by reading, not run on hardware
//
#include <hip/hip_runtime.h>
#include <math.h>

// ---------------- problem constants (from reference) ----------------
constexpr int BATCH   = 2;
constexpr int SEQ     = 2048;
constexpr int CHN     = 512;            // channels (= K dim of every GEMM)
constexpr int NHEAD   = 8;
constexpr int HDIM    = CHN / NHEAD;    // 64
constexpr int KTAP    = 64;             // kernel taps allocated
constexpr int HALF_S  = 16;
constexpr int STAP    = 2 * HALF_S + 1; // 33 taps used
constexpr float MAX_F = 16.0f;
constexpr float MIN_F = 1.0f;
constexpr float MAX_RECEPTIVE = HALF_S * MAX_F; // 256
constexpr int MROWS   = BATCH * SEQ;    // 4096
constexpr float LOG2E = 1.44269504088896340736f;

typedef __bf16 v16bf __attribute__((ext_vector_type(16)));
typedef __bf16 v8bf  __attribute__((ext_vector_type(8)));
typedef float  v8f   __attribute__((ext_vector_type(8)));
typedef float  v4f   __attribute__((ext_vector_type(4)));

// Branchless fast transcendentals on the hardware TRANS pipe
// (v_exp_f32 + v_rcp_f32) -- no IEEE divide chain, no EXEC divergence.
__device__ __forceinline__ float fast_sigmoid(float v) {
    return __builtin_amdgcn_rcpf(1.0f + __builtin_amdgcn_exp2f(-LOG2E * v));
}
__device__ __forceinline__ float fast_silu(float v) {
    return v * fast_sigmoid(v);
}
__device__ __forceinline__ float fast_tanh(float v) {
    // tanh(v) = 1 - 2/(exp(2v)+1); exp2 saturates to +inf / 0 -> tanh -> +/-1
    return 1.0f - 2.0f * __builtin_amdgcn_rcpf(
        1.0f + __builtin_amdgcn_exp2f(2.0f * LOG2E * v));
}

// --------------------------------------------------------------------
// Streaming fp32 -> bf16 pre-pack (one pass; amortized over 256-512
// reuses of each operand inside the GEMMs).
// --------------------------------------------------------------------
__global__ __launch_bounds__(256)
void cvt_f32_to_bf16(const float* __restrict__ in, __bf16* __restrict__ out,
                     int nvec /* = n/8 */)
{
    for (int i = blockIdx.x * blockDim.x + threadIdx.x; i < nvec;
         i += gridDim.x * blockDim.x) {
        v4f a = *(const v4f*)(in + (size_t)i * 8);
        v4f b = *(const v4f*)(in + (size_t)i * 8 + 4);
        v8bf o;
#pragma unroll
        for (int j = 0; j < 4; ++j) {
            o[j]     = (__bf16)a[j];
            o[j + 4] = (__bf16)b[j];
        }
        *(v8bf*)(out + (size_t)i * 8) = o;
    }
}

// --------------------------------------------------------------------
// WMMA GEMM:  D[m,n] = epilogue( sum_k A[m,k] * W[n,k] + bias[n] )
// A: [M,512] bf16 row-major, W: [N,512] bf16 row-major (B = W^T).
// One wave computes a 16 x (16*NT) strip: the A fragment is loaded once
// per k-step and fed to NT independent back-to-back WMMAs.
//
// Fragment packing per ISA 7.12.2 (16-bit A, 16x32):
//   lanes 0-15 : M=lane,    elems 0..7 -> K=k0..k0+7,    8..15 -> K=k0+16..+23
//   lanes 16-31: M=lane-16, elems 0..7 -> K=k0+8..k0+15, 8..15 -> K=k0+24..+31
// B fragment packed symmetrically with N in the lane slot.
// --------------------------------------------------------------------
template <int NT, bool WAVE_EPI>
__global__ __launch_bounds__(256)
void wmma_gemm_bf16(const __bf16* __restrict__ A,
                    const __bf16* __restrict__ W,
                    const float* __restrict__ bias,
                    float* __restrict__ out0,   // STD: C matrix | WAVE: freq
                    float* __restrict__ out1,   // WAVE: phase
                    int M, int N)
{
    constexpr int KDIM = CHN; // 512 for every GEMM in this net
    const int wave = threadIdx.x >> 5;
    const int lane = threadIdx.x & 31;
    const int stripsN = N / (16 * NT);
    const int strip = blockIdx.x * 8 + wave;
    if (strip >= (M >> 4) * stripsN) return;

    const int tm = strip / stripsN;
    const int tn = strip % stripsN;

    const int lrow  = lane & 15;
    const int khalf = (lane & 16) >> 1;  // 0 for lanes 0-15, 8 for lanes 16-31

    const __bf16* __restrict__ arow =
        A + (size_t)(tm * 16 + lrow) * KDIM + khalf;
    const __bf16* __restrict__ brow =
        W + (size_t)(tn * (16 * NT) + lrow) * KDIM + khalf;

    v8f acc[NT];
#pragma unroll
    for (int t = 0; t < NT; ++t) acc[t] = (v8f){};

#pragma unroll 2
    for (int k0 = 0; k0 < KDIM; k0 += 32) {
        // A fragment: two 16B bf16 loads, concatenated
        v8bf alo = *(const v8bf*)(arow + k0);
        v8bf ahi = *(const v8bf*)(arow + k0 + 16);
        v16bf af = __builtin_shufflevector(alo, ahi,
            0, 1, 2, 3, 4, 5, 6, 7, 8, 9, 10, 11, 12, 13, 14, 15);

        v16bf bfv[NT];
#pragma unroll
        for (int t = 0; t < NT; ++t) {
            const __bf16* bp = brow + (size_t)t * 16 * KDIM + k0;
            v8bf blo = *(const v8bf*)(bp);
            v8bf bhi = *(const v8bf*)(bp + 16);
            bfv[t] = __builtin_shufflevector(blo, bhi,
                0, 1, 2, 3, 4, 5, 6, 7, 8, 9, 10, 11, 12, 13, 14, 15);
        }
#pragma unroll
        for (int t = 0; t < NT; ++t) {
            acc[t] = __builtin_amdgcn_wmma_f32_16x16x32_bf16(
                false, af, false, bfv[t], (short)0, acc[t], false, false);
        }
    }

    // C/D layout: VGPR r -> M = r (lanes 0-15) or r+8 (lanes 16-31); N = lane&15
    const int mbase = tm * 16 + khalf;
#pragma unroll
    for (int t = 0; t < NT; ++t) {
        const int n  = tn * (16 * NT) + t * 16 + lrow;
        const float bv = bias ? bias[n] : 0.0f;
#pragma unroll
        for (int r = 0; r < 8; ++r) {
            float v = fast_silu(acc[t][r] + bv);
            if (WAVE_EPI) {
                // N==16: cols 0-7 -> freq, 8-15 -> phase
                const int row = mbase + r;
                if (n < NHEAD)
                    out0[row * NHEAD + n] = fast_sigmoid(v) * (MAX_F - MIN_F) + MIN_F;
                else
                    out1[row * NHEAD + (n - NHEAD)] = fast_tanh(v) * MAX_F;
            } else {
                out0[(size_t)(mbase + r) * N + n] = v;
            }
        }
    }
}

// --------------------------------------------------------------------
// Gather convolution. One block per (b,l) query position.
// hidden[b,l,h,d] = sum_s kern[b,l,h,s] * lerp(x[b, pos(b,l,h,s), h, d])
// Taps + freq/phase staged through LDS; each thread handles two (h,d)
// channels so consecutive threads touch consecutive addresses.
// Output written as bf16 (it feeds the bf16 WMMA output GEMM).
// --------------------------------------------------------------------
__global__ __launch_bounds__(256)
void gather_conv_kernel(const float* __restrict__ x,
                        const float* __restrict__ freq,
                        const float* __restrict__ phase,
                        const float* __restrict__ kern,
                        __bf16* __restrict__ hidden)
{
    __shared__ float s_kern[NHEAD * STAP];
    __shared__ float s_freq[NHEAD];
    __shared__ float s_phase[NHEAD];

    const int row = blockIdx.x;        // b*SEQ + l
    const int b   = row >> 11;         // SEQ = 2048
    const int l   = row & (SEQ - 1);
    const int tid = threadIdx.x;

    for (int t = tid; t < NHEAD * STAP; t += 256) {
        const int h = t / STAP;
        const int s = t - h * STAP;
        s_kern[t] = kern[(size_t)row * (NHEAD * KTAP) + h * KTAP + s];
    }
    if (tid < NHEAD) {
        s_freq[tid]  = freq[row * NHEAD + tid];
        s_phase[tid] = phase[row * NHEAD + tid];
    }
    __syncthreads();

    const float* __restrict__ xb = x + (size_t)b * SEQ * CHN;
    const float tl  = (float)l;
    const float lo1 = tl - MAX_RECEPTIVE;
    const float hi1 = tl + MAX_RECEPTIVE;

#pragma unroll
    for (int e0 = 0; e0 < 2; ++e0) {
        const int e = tid + e0 * 256;       // channel index = h*HDIM + d
        const int h = e >> 6;               // HDIM = 64
        const float fr = s_freq[h];
        const float ph = s_phase[h];
        float acc = 0.0f;
        for (int s = 0; s < STAP; ++s) {
            float pos = tl + ph + (float)(s - HALF_S) * fr;
            pos = fminf(fmaxf(pos, lo1), hi1);
            pos = fminf(fmaxf(pos, 0.0f), (float)(SEQ - 1));
            const float p0 = floorf(pos);
            const float w  = pos - p0;
            const int i0 = (int)p0;
            const int i1 = min(i0 + 1, SEQ - 1);
            const float g0 = xb[i0 * CHN + e];
            const float g1 = xb[i1 * CHN + e];
            acc = fmaf(s_kern[h * STAP + s], (1.0f - w) * g0 + w * g1, acc);
        }
        hidden[(size_t)row * CHN + e] = (__bf16)acc;
    }
}

// --------------------------------------------------------------------
extern "C" void kernel_launch(void* const* d_in, const int* in_sizes, int n_in,
                              void* d_out, int out_size, void* d_ws, size_t ws_size,
                              hipStream_t stream)
{
    (void)in_sizes; (void)n_in; (void)out_size; (void)ws_size;

    const float* x        = (const float*)d_in[0];
    const float* wave_w   = (const float*)d_in[1];
    const float* wave_b   = (const float*)d_in[2];
    const float* kernel_w = (const float*)d_in[3];
    const float* kernel_b = (const float*)d_in[4];
    const float* out_w    = (const float*)d_in[5];
    float* out = (float*)d_out;

    // ---- workspace layout ----
    // f32:  freq | phase | kern            (4096*8, 4096*8, 4096*512)
    // bf16: x_bf | wavew_bf | kernw_bf | outw_bf | hidden_bf
    float* wsf    = (float*)d_ws;
    float* freq   = wsf;                                  // 32768 f
    float* phase  = freq  + (size_t)MROWS * NHEAD;        // 32768 f
    float* kern   = phase + (size_t)MROWS * NHEAD;        // 2097152 f
    __bf16* x_bf     = (__bf16*)(kern + (size_t)MROWS * CHN);
    __bf16* wavew_bf = x_bf     + (size_t)MROWS * CHN;    // 16*512
    __bf16* kernw_bf = wavew_bf + (size_t)16 * CHN;       // 512*512
    __bf16* outw_bf  = kernw_bf + (size_t)CHN * CHN;      // 512*512
    __bf16* hidden_bf= outw_bf  + (size_t)CHN * CHN;      // 4096*512

    // 0) one-time bf16 pre-pack of GEMM operands
    cvt_f32_to_bf16<<<1024, 256, 0, stream>>>(x, x_bf, MROWS * CHN / 8);
    cvt_f32_to_bf16<<<4,    256, 0, stream>>>(wave_w, wavew_bf, 16 * CHN / 8);
    cvt_f32_to_bf16<<<128,  256, 0, stream>>>(kernel_w, kernw_bf, CHN * CHN / 8);
    cvt_f32_to_bf16<<<128,  256, 0, stream>>>(out_w, outw_bf, CHN * CHN / 8);

    // 1) wave GEMM (M=4096, N=16, K=512) with fused freq/phase epilogue
    {
        const int strips = (MROWS / 16) * (16 / 16);      // 256
        wmma_gemm_bf16<1, true><<<strips / 8, 256, 0, stream>>>(
            x_bf, wavew_bf, wave_b, freq, phase, MROWS, 16);
    }
    // 2) kernel taps GEMM (M=4096, N=512, K=512), silu epilogue -> f32 kern
    {
        const int strips = (MROWS / 16) * (CHN / 64);     // 2048
        wmma_gemm_bf16<4, false><<<strips / 8, 256, 0, stream>>>(
            x_bf, kernw_bf, kernel_b, kern, nullptr, MROWS, CHN);
    }
    // 3) learned-frequency gather convolution (f32 math, bf16 output)
    gather_conv_kernel<<<MROWS, 256, 0, stream>>>(x, freq, phase, kern, hidden_bf);

    // 4) output GEMM (M=4096, N=512, K=512), silu epilogue, no bias -> f32 out
    {
        const int strips = (MROWS / 16) * (CHN / 64);     // 2048
        wmma_gemm_bf16<4, false><<<strips / 8, 256, 0, stream>>>(
            hidden_bf, outw_bf, nullptr, out, nullptr, MROWS, CHN);
    }
}
